// RNN_53867479827170
// MI455X (gfx1250) — compile-verified, matrix operand order checked by reading, not run on hardware
//
#include <hip/hip_runtime.h>

// LSTM on MI455X (gfx1250): per-timestep fused GEMM (bf16 WMMA, f32 accum) +
// cell update via hardware v_tanh_f32. Weights transposed/converted once per
// launch into workspace and stay L2-resident (12 MB bf16 << 192 MB L2) across
// all 1024 steps.

typedef __attribute__((ext_vector_type(16))) __bf16 v16bf;
typedef __attribute__((ext_vector_type(8)))  float  v8f;

#define B_N 64
#define T_N 1024
#define D_N 512
#define H_N 1024
#define K_N 1536            // D + H
#define A_PITCH 1544        // K + 8 elem pad, keeps rows 16B-aligned

// ---- gfx1250 hardware tanh (TRANS op). Builtin preferred so the compiler
// ---- handles the TRANS co-execution hazard; asm fallback adds a v_nop.
__device__ __forceinline__ float dtanh(float x) {
#if __has_builtin(__builtin_amdgcn_tanhf)
    return __builtin_amdgcn_tanhf(x);
#elif defined(__gfx1250__)
    float r;
    asm("v_tanh_f32 %0, %1\n\tv_nop" : "=v"(r) : "v"(x));
    return r;
#else
    return tanhf(x);
#endif
}
__device__ __forceinline__ float dsigmoid(float x) {
    return fmaf(0.5f, dtanh(0.5f * x), 0.5f);
}

// ---------------- weight prep: W_t[n][k] = (k<D ? Wx[k][n] : Wh[k-D][n]) as bf16
__global__ void prep_weights(const float* __restrict__ Wx,
                             const float* __restrict__ Wh,
                             __bf16* __restrict__ wt) {
    int gid = blockIdx.x * blockDim.x + threadIdx.x;   // over 4096*1536
    int col = gid / K_N;
    int k   = gid - col * K_N;
    if (col >= 4 * H_N) return;
    float v = (k < D_N) ? Wx[(size_t)k * (4 * H_N) + col]
                        : Wh[(size_t)(k - D_N) * (4 * H_N) + col];
    wt[(size_t)col * K_N + k] = (__bf16)v;
}

// ---------------- zero h (both ping-pong buffers, bf16) and c (f32)
__global__ void init_state(__bf16* __restrict__ h0, __bf16* __restrict__ h1,
                           float* __restrict__ c) {
    int gid = blockIdx.x * blockDim.x + threadIdx.x;
    if (gid < B_N * H_N) {
        h0[gid] = (__bf16)0.f;
        h1[gid] = (__bf16)0.f;
        c[gid]  = 0.f;
    }
}

union FragU {
    uint4 q[2];
    v16bf v;
};

// One timestep. grid = (64 n-blocks, 4 m-tiles), 128 threads = 4 waves.
// Wave g computes the 16x16 tile of gate g (columns g*H + nb*16 .. +15).
__launch_bounds__(128, 1)
__global__ void lstm_step(const float*  __restrict__ x,      // [B,T,D] f32
                          const float*  __restrict__ bias,   // [4H] f32
                          const __bf16* __restrict__ wt,     // [4H][K] bf16 (N-major)
                          const __bf16* __restrict__ hprev,  // [B,H] bf16
                          __bf16*       __restrict__ hnext,  // [B,H] bf16
                          float*        __restrict__ cstate, // [B,H] f32
                          float*        __restrict__ out,    // [B,T,H] f32
                          int t) {
    __shared__ __align__(16) __bf16 sA[16 * A_PITCH];  // A panel: [x_t | h_{t-1}], 16 rows
    __shared__ float sG[4][16][17];                    // gate exchange (f32 accum)

    const int tid  = threadIdx.x;
    const int lane = tid & 31;
    const int wave = tid >> 5;       // gate id 0..3
    const int nb   = blockIdx.x;     // 16-column block within H
    const int m0   = blockIdx.y * 16;

    // ---- stage A into LDS: x part (f32 -> bf16), 128 float4 per row ----
    for (int idx = tid; idx < 16 * (D_N / 4); idx += 128) {
        int r  = idx >> 7;
        int k4 = (idx & 127) << 2;
        const float4 v = *(const float4*)(x + (size_t)(m0 + r) * T_N * D_N
                                            + (size_t)t * D_N + k4);
        __bf16* dst = sA + r * A_PITCH + k4;
        dst[0] = (__bf16)v.x; dst[1] = (__bf16)v.y;
        dst[2] = (__bf16)v.z; dst[3] = (__bf16)v.w;
    }
    // ---- h part: raw bf16 copy, 128 x 16B per row ----
    for (int idx = tid; idx < 16 * (H_N / 8); idx += 128) {
        int r  = idx >> 7;
        int k8 = (idx & 127) << 3;
        uint4 v = *(const uint4*)(hprev + (size_t)(m0 + r) * H_N + k8);
        *(uint4*)(sA + r * A_PITCH + D_N + k8) = v;
    }

    // ---- 16x16x(K=1536) GEMM tile per wave ----
    const int nLane = lane & 15;          // A row (M) / B column (N)
    const int khalf = lane >> 4;          // which K half-group this lane holds
    const int colBase = wave * H_N + nb * 16;
    const __bf16* wrow = wt + (size_t)(colBase + nLane) * K_N;
    const __bf16* arow = sA + nLane * A_PITCH;

    // Fold bias into the accumulator: C/D layout has column == lane's nLane
    // for every accumulator VGPR, so one scalar load seeds all 8 regs.
    const float bval = bias[colBase + nLane];
    v8f acc;
    #pragma unroll
    for (int j = 0; j < 8; ++j) acc[j] = bval;

    __syncthreads();

    #pragma unroll 4
    for (int kb = 0; kb < K_N / 32; ++kb) {
        const int k0 = kb * 32;
        FragU a, b;
        // bf16 16x32 fragment layout: lanes 0-15 hold K 0-7 & 16-23,
        // lanes 16-31 hold K 8-15 & 24-31 (two 16B chunks per lane).
        a.q[0] = *(const uint4*)(arow + k0 + khalf * 8);
        a.q[1] = *(const uint4*)(arow + k0 + 16 + khalf * 8);
        b.q[0] = *(const uint4*)(wrow + k0 + khalf * 8);
        b.q[1] = *(const uint4*)(wrow + k0 + 16 + khalf * 8);
        acc = __builtin_amdgcn_wmma_f32_16x16x32_bf16(false, a.v, false, b.v,
                                                      (short)0, acc, false, false);
    }

    // ---- dump f32 accumulators to LDS (C/D layout: vgpr j -> row j or j+8) ----
    {
        int row0 = khalf << 3;
        #pragma unroll
        for (int j = 0; j < 8; ++j)
            sG[wave][row0 + j][nLane] = acc[j];
    }
    __syncthreads();

    // ---- fused cell update: 256 elements, 2 per thread ----
    for (int e = tid; e < 256; e += 128) {
        int r  = e >> 4;
        int nl = e & 15;
        int n  = nb * 16 + nl;
        int m  = m0 + r;
        float si = dsigmoid(sG[0][r][nl]);
        float sf = dsigmoid(sG[1][r][nl]);
        float tg = dtanh(sG[2][r][nl]);
        float so = dsigmoid(sG[3][r][nl]);
        float cn = fmaf(sf, cstate[m * H_N + n], si * tg);
        float hn = so * dtanh(cn);
        cstate[m * H_N + n] = cn;
        hnext[m * H_N + n]  = (__bf16)hn;
        out[(size_t)m * T_N * H_N + (size_t)t * H_N + n] = hn;
    }
}

extern "C" void kernel_launch(void* const* d_in, const int* in_sizes, int n_in,
                              void* d_out, int out_size, void* d_ws, size_t ws_size,
                              hipStream_t stream) {
    const float* x  = (const float*)d_in[0];   // [B,T,D]
    const float* Wx = (const float*)d_in[1];   // [D,4H]
    const float* Wh = (const float*)d_in[2];   // [H,4H]
    const float* b  = (const float*)d_in[3];   // [4H]
    float* out = (float*)d_out;                // [B,T,H]

    // workspace layout (~13 MB total):
    //   wt : bf16 [4096][1536]   12 MB
    //   h0 : bf16 [64][1024]     128 KB  (ping)
    //   h1 : bf16 [64][1024]     128 KB  (pong)
    //   c  : f32  [64][1024]     256 KB
    char* ws = (char*)d_ws;
    __bf16* wt = (__bf16*)ws;
    __bf16* h0 = (__bf16*)(ws + (size_t)4 * H_N * K_N * 2);
    __bf16* h1 = h0 + B_N * H_N;
    float*  c  = (float*)(ws + (size_t)4 * H_N * K_N * 2 + (size_t)2 * B_N * H_N * 2);

    {
        int total = 4 * H_N * K_N;
        prep_weights<<<(total + 255) / 256, 256, 0, stream>>>(Wx, Wh, wt);
    }
    init_state<<<(B_N * H_N + 255) / 256, 256, 0, stream>>>(h0, h1, c);

    dim3 grid(64, 4);
    for (int t = 0; t < T_N; ++t) {
        const __bf16* hp = (t & 1) ? h1 : h0;
        __bf16*       hn = (t & 1) ? h0 : h1;
        lstm_step<<<grid, 128, 0, stream>>>(x, b, wt, hp, hn, c, out, t);
    }
}
